// InteractionLayer_24592982736979
// MI455X (gfx1250) — matrix operand order repaired and checked by reading, not compiled, not run on hardware
//
#include <hip/hip_runtime.h>

#define V_NODES 50000
#define E_EDGES 600000
#define DH 128

typedef _Float16 v16h __attribute__((ext_vector_type(16)));
typedef _Float16 v8h  __attribute__((ext_vector_type(8)));
typedef float    v8f  __attribute__((ext_vector_type(8)));

// ---- WMMA fragment loaders (CDNA5 wave32 layouts, CDNA5 ISA §7.12.2) ----

// A (16x32 f16): lane l<16 holds row l, K = kb+{0..7} and kb+{16..23};
// lane l>=16 holds row l-16, K = kb+{8..15} and kb+{24..31}.  Row-major LDS tile [rows][128].
__device__ __forceinline__ v16h load_a_frag(const _Float16* lds, int lane, int kb) {
    const int m  = lane & 15;
    const int hi = (lane >> 4) << 3;                  // 0 or 8
    const _Float16* p = lds + m * 128 + kb + hi;
    v8h lo = *(const v8h*)(p);
    v8h up = *(const v8h*)(p + 16);
    return __builtin_shufflevector(lo, up, 0,1,2,3,4,5,6,7,8,9,10,11,12,13,14,15);
}

// Per-lane B base: fragment (t,kb) then lives at base + t*2048 + kb (halves),
// a compile-time offset after unrolling -> single address VGPR pair, imm offsets.
__device__ __forceinline__ const _Float16* b_base(const _Float16* __restrict__ W, int lane) {
    return W + (lane & 15) * 128 + ((lane >> 4) << 3);
}
__device__ __forceinline__ v16h load_b_at(const _Float16* base, int off) {
    v8h lo = *(const v8h*)(base + off);
    v8h up = *(const v8h*)(base + off + 16);
    return __builtin_shufflevector(lo, up, 0,1,2,3,4,5,6,7,8,9,10,11,12,13,14,15);
}
__device__ __forceinline__ constexpr int b_off(int i) {   // i = kb-step*8 + t
    return (i & 7) * 2048 + ((i >> 3) * 32);
}

// [32x128] = A(32x128 in LDS, f16) x W^T(128x128 f16), f32 accumulate.
// Two M-tiles share every B fragment; B double-buffered so the next fragment is
// in flight while the current WMMA pair executes.
__device__ __forceinline__ void gemm_32x128(const _Float16* aLds,
                                            const _Float16* __restrict__ W,
                                            v8f acc0[8], v8f acc1[8], int lane) {
    const _Float16* base = b_base(W, lane);
    v16h a0 = load_a_frag(aLds,            lane, 0);
    v16h a1 = load_a_frag(aLds + 16 * 128, lane, 0);
    v16h b  = load_b_at(base, 0);
#pragma unroll
    for (int i = 0; i < 32; ++i) {
        const int kb = (i >> 3) * 32;
        const int t  = i & 7;
        v16h bn = b;
        if (i < 31) bn = load_b_at(base, b_off(i + 1));
        acc0[t] = __builtin_amdgcn_wmma_f32_16x16x32_f16(
            false, a0, false, b, (short)0, acc0[t], false, false);
        acc1[t] = __builtin_amdgcn_wmma_f32_16x16x32_f16(
            false, a1, false, b, (short)0, acc1[t], false, false);
        if (t == 7 && kb < 96) {
            a0 = load_a_frag(aLds,            lane, kb + 32);
            a1 = load_a_frag(aLds + 16 * 128, lane, kb + 32);
        }
        b = bn;
    }
}

// [16x128] variant for the node-update kernel, same structure.
__device__ __forceinline__ void gemm_16x128(const _Float16* aLds,
                                            const _Float16* __restrict__ W,
                                            v8f acc[8], int lane) {
    const _Float16* base = b_base(W, lane);
    v16h a = load_a_frag(aLds, lane, 0);
    v16h b = load_b_at(base, 0);
#pragma unroll
    for (int i = 0; i < 32; ++i) {
        const int kb = (i >> 3) * 32;
        const int t  = i & 7;
        v16h bn = b;
        if (i < 31) bn = load_b_at(base, b_off(i + 1));
        acc[t] = __builtin_amdgcn_wmma_f32_16x16x32_f16(
            false, a, false, b, (short)0, acc[t], false, false);
        if (t == 7 && kb < 96) a = load_a_frag(aLds, lane, kb + 32);
        b = bn;
    }
}

// ---- weight f32 -> f16 conversion (F1,F2,U1,U2 -> ws) ----
__global__ __launch_bounds__(256) void cvt_kernel(const float* __restrict__ F1,
                                                  const float* __restrict__ F2,
                                                  const float* __restrict__ U1,
                                                  const float* __restrict__ U2,
                                                  _Float16* __restrict__ dst) {
    int i = blockIdx.x * 256 + threadIdx.x;           // 0 .. 4*16384-1
    const float* srcs[4] = {F1, F2, U1, U2};
    dst[i] = (_Float16)srcs[i >> 14][i & 16383];
}

// ---- fused edge kernel: dist -> RBF -> relu(relu(rbf@F1^T)@F2^T) * nf[src] -> atomic H[dest]
// 4 waves/block, each wave owns a 32-edge tile.
__global__ __launch_bounds__(128) void edge_kernel(const float* __restrict__ node_feats,
                                                   const float* __restrict__ coords,
                                                   const int*   __restrict__ src,
                                                   const int*   __restrict__ dest,
                                                   const float* __restrict__ centers,
                                                   const _Float16* __restrict__ F1h,
                                                   const _Float16* __restrict__ F2h,
                                                   float* __restrict__ H) {
    __shared__ __align__(16) _Float16 s_a[4][32 * 128];   // 32 KB act/rbf staging
    __shared__ int s_src[4][32];
    __shared__ int s_dst[4][32];

    const int w    = threadIdx.x >> 5;
    const int lane = threadIdx.x & 31;
    const int tile = blockIdx.x * 4 + w;
    if (tile >= E_EDGES / 32) return;                 // terminated waves satisfy barriers
    const int ebase = tile * 32;

    // one edge per lane: distance stays in-register, indices go to LDS for the epilogue
    float dist;
    {
        int e = ebase + lane;
        int s = src[e], d = dest[e];
        float dx = coords[3 * s + 0] - coords[3 * d + 0];
        float dy = coords[3 * s + 1] - coords[3 * d + 1];
        float dz = coords[3 * s + 2] - coords[3 * d + 2];
        dist = sqrtf(dx * dx + dy * dy + dz * dz);
        s_src[w][lane] = s;
        s_dst[w][lane] = d;
    }

    // RBF row (128 bases) per lane, packed to b128 LDS stores
    {
        const float delta = 5.0f / 127.0f;
        const float gamma = 1.0f / (2.0f * delta * delta);
        _Float16* row = &s_a[w][lane * 128];
#pragma unroll
        for (int k8 = 0; k8 < 16; ++k8) {
            v8h pack;
#pragma unroll
            for (int j = 0; j < 8; ++j) {
                float diff = dist - centers[k8 * 8 + j];
                pack[j] = (_Float16)__expf(-gamma * diff * diff);
            }
            *(v8h*)(row + k8 * 8) = pack;
        }
    }
    __syncthreads();

    v8f acc0[8] = {}, acc1[8] = {};
    gemm_32x128(s_a[w], F1h, acc0, acc1, lane);
    __syncthreads();

    const int n0 = lane & 15;
    const int mh = (lane >> 4) * 8;                   // C/D layout: elem r -> row r+mh
#pragma unroll
    for (int t = 0; t < 8; ++t)
#pragma unroll
        for (int r = 0; r < 8; ++r) {
            float v0 = acc0[t][r];
            float v1 = acc1[t][r];
            s_a[w][(mh + r) * 128 + t * 16 + n0]        = (_Float16)(v0 > 0.0f ? v0 : 0.0f);
            s_a[w][(16 + mh + r) * 128 + t * 16 + n0]   = (_Float16)(v1 > 0.0f ? v1 : 0.0f);
        }
    __syncthreads();

    v8f m0[8] = {}, m1[8] = {};
    gemm_32x128(s_a[w], F2h, m0, m1, lane);

#pragma unroll
    for (int half = 0; half < 2; ++half) {
#pragma unroll
        for (int r = 0; r < 8; ++r) {
            const int m = half * 16 + mh + r;
            const float* nf = node_feats + (size_t)s_src[w][m] * DH;
            float*       hp = H          + (size_t)s_dst[w][m] * DH;
#pragma unroll
            for (int t = 0; t < 8; ++t) {
                float v = half ? m1[t][r] : m0[t][r];
                v = (v > 0.0f ? v : 0.0f) * nf[t * 16 + n0];
                atomicAdd(hp + t * 16 + n0, v);
            }
        }
    }
}

// ---- node update: out = relu(H@U1^T + b1) @ U2^T + b2  (M=16 per wave) ----
__global__ __launch_bounds__(256) void update_kernel(const float* __restrict__ Hin,
                                                     const _Float16* __restrict__ U1h,
                                                     const _Float16* __restrict__ U2h,
                                                     const float* __restrict__ Ub1,
                                                     const float* __restrict__ Ub2,
                                                     float* __restrict__ out) {
    __shared__ __align__(16) _Float16 s_a[8][16 * 128];
    const int w    = threadIdx.x >> 5;
    const int lane = threadIdx.x & 31;
    const int tile = blockIdx.x * 8 + w;
    if (tile >= V_NODES / 16) return;
    const int vbase = tile * 16;

    {   // load H tile -> f16 LDS
        const int m  = lane & 15;
        const int k0 = (lane >> 4) * 64;
        const float* hp = Hin + (size_t)(vbase + m) * DH + k0;
#pragma unroll
        for (int k = 0; k < 64; ++k) s_a[w][m * 128 + k0 + k] = (_Float16)hp[k];
    }
    __syncthreads();

    v8f acc[8] = {};
    gemm_16x128(s_a[w], U1h, acc, lane);
    __syncthreads();

    const int n0 = lane & 15;
    const int mh = (lane >> 4) * 8;
#pragma unroll
    for (int t = 0; t < 8; ++t) {
        float b = Ub1[t * 16 + n0];
#pragma unroll
        for (int r = 0; r < 8; ++r) {
            float v = acc[t][r] + b;
            s_a[w][(mh + r) * 128 + t * 16 + n0] = (_Float16)(v > 0.0f ? v : 0.0f);
        }
    }
    __syncthreads();

    v8f acc2[8] = {};
    gemm_16x128(s_a[w], U2h, acc2, lane);

#pragma unroll
    for (int t = 0; t < 8; ++t) {
        float b = Ub2[t * 16 + n0];
#pragma unroll
        for (int r = 0; r < 8; ++r)
            out[(size_t)(vbase + mh + r) * DH + t * 16 + n0] = acc2[t][r] + b;
    }
}

extern "C" void kernel_launch(void* const* d_in, const int* in_sizes, int n_in,
                              void* d_out, int out_size, void* d_ws, size_t ws_size,
                              hipStream_t stream) {
    const float* node_feats = (const float*)d_in[0];
    const float* coords     = (const float*)d_in[1];
    // d_in[2] batch_index: unused
    const int*   src        = (const int*)d_in[3];
    const int*   dest       = (const int*)d_in[4];
    // d_in[5] W_w, d_in[6] W_b: dead in reference
    const float* centers    = (const float*)d_in[7];
    const float* F1         = (const float*)d_in[8];
    const float* F2         = (const float*)d_in[9];
    const float* U1         = (const float*)d_in[10];
    const float* Ub1        = (const float*)d_in[11];
    const float* U2         = (const float*)d_in[12];
    const float* Ub2        = (const float*)d_in[13];
    float* out = (float*)d_out;

    // workspace: H [V,128] f32, then 4x 128x128 f16 weights
    float*    H  = (float*)d_ws;
    _Float16* Wh = (_Float16*)((char*)d_ws + (size_t)V_NODES * DH * sizeof(float));
    _Float16* F1h = Wh;
    _Float16* F2h = Wh + 16384;
    _Float16* U1h = Wh + 32768;
    _Float16* U2h = Wh + 49152;

    hipMemsetAsync(H, 0, (size_t)V_NODES * DH * sizeof(float), stream);
    cvt_kernel<<<256, 256, 0, stream>>>(F1, F2, U1, U2, Wh);

    const int etiles = E_EDGES / 32;                  // 18750 exact
    edge_kernel<<<(etiles + 3) / 4, 128, 0, stream>>>(node_feats, coords, src, dest,
                                                      centers, F1h, F2h, H);
    const int vtiles = V_NODES / 16;                  // 3125 exact
    update_kernel<<<(vtiles + 7) / 8, 256, 0, stream>>>(H, U1h, U2h, Ub1, Ub2, out);
}
